// LaBraMEncoderV2_83648783057669
// MI455X (gfx1250) — matrix operand order repaired and checked by reading, not compiled, not run on hardware
//
#include <hip/hip_runtime.h>
#include <hip/hip_bf16.h>
#include <math.h>

// ---------------------------------------------------------------------------
// LaBraM-style encoder on MI455X (gfx1250, wave32, WMMA).
// All contractions run through v_wmma_f32_16x16x32_f16 (f16 A/B, f32 acc).
// Conv stack fully fused per 16-sequence workgroup; intermediates live in LDS
// with per-sequence zero padding so conv fragments need no predication.
// ---------------------------------------------------------------------------

typedef _Float16 h16;
typedef __attribute__((ext_vector_type(16))) _Float16 v16h;
typedef __attribute__((ext_vector_type(8)))  float    v8f;

struct alignas(16) U128 { unsigned int a, b, c, d; };

union V16U { v16h v; U128 q[2]; };

__device__ inline float gelu_f(float x) {
    // exact (erf) GELU, matches jax.nn.gelu(approximate=False)
    return 0.5f * x * (1.0f + erff(x * 0.70710678118654752f));
}

// D = A x B + C, 16x16x32 f16 -> f32
__device__ inline v8f wmma_f16(v16h a, v16h b, v8f c) {
    return __builtin_amdgcn_wmma_f32_16x16x32_f16(
        false, a, false, b, (short)0, c, false, false);
}

// A-matrix fragment: A[M][K] row-major (ld halves per row).
// lane row = mbase + (lane&15); element e -> K = kbase + 16*(lane/16) + e
__device__ inline v16h frag_a(const h16* __restrict__ base, int ld, int mbase, int kbase) {
    int lane = threadIdx.x & 31;
    const h16* p = base + (mbase + (lane & 15)) * ld + kbase + ((lane >> 4) << 4);
    V16U f;
    f.q[0] = *(const U128*)(p);
    f.q[1] = *(const U128*)(p + 8);
    return f.v;
}

// B-matrix fragment from B^T stored as BT[N][K] row-major.
// lane col = nbase + (lane&15); element e -> K = kbase + 16*(lane/16) + e (contiguous 32B)
__device__ inline v16h frag_bt(const h16* __restrict__ base, int ld, int nbase, int kbase) {
    int lane = threadIdx.x & 31;
    const h16* p = base + (nbase + (lane & 15)) * ld + kbase + ((lane >> 4) << 4);
    V16U f;
    f.q[0] = *(const U128*)(p);
    f.q[1] = *(const U128*)(p + 8);
    return f.v;
}

// Raw contiguous 32B fragment load (per-lane pointer already resolved).
__device__ inline v16h frag_ld(const h16* p) {
    V16U f;
    f.q[0] = *(const U128*)(p);
    f.q[1] = *(const U128*)(p + 8);
    return f.v;
}

// ---------------------------------------------------------------------------
// Kernel 0: weight conversion / re-layout to f16 (padded, fragment friendly)
//   w2h [co=128][tap=3][ci=64]       24576 h
//   w3h [co=256][tap=3][ci=128]      98304 h
//   mixh[o=144 pad][c=160 pad]       23040 h (zero padded)
//   projh[e=256][d=256]              65536 h
// ---------------------------------------------------------------------------
__global__ __launch_bounds__(256)
void prep_kernel(const float* __restrict__ w2, const float* __restrict__ w3,
                 const float* __restrict__ mix_w, const float* __restrict__ proj_w,
                 h16* __restrict__ w2h, h16* __restrict__ w3h,
                 h16* __restrict__ mixh, h16* __restrict__ projh) {
    int i = blockIdx.x * 256 + threadIdx.x;
    if (i < 24576) {
        int co = i / 192, rem = i % 192, tap = rem >> 6, ci = rem & 63;
        w2h[i] = (h16)w2[(co * 64 + ci) * 3 + tap];
    }
    int j = i - 24576;
    if (j >= 0 && j < 98304) {
        int co = j / 384, rem = j % 384, tap = rem >> 7, ci = rem & 127;
        w3h[j] = (h16)w3[(co * 128 + ci) * 3 + tap];
    }
    int k = i - (24576 + 98304);
    if (k >= 0 && k < 23040) {
        int o = k / 160, c = k % 160;
        mixh[k] = (h16)((o < 129 && c < 129) ? mix_w[o * 129 + c] : 0.0f);
    }
    int l = i - (24576 + 98304 + 23040);
    if (l >= 0 && l < 65536) {
        projh[l] = (h16)proj_w[l];   // already [e][d] row-major
    }
}

// ---------------------------------------------------------------------------
// Kernel 1: fused conv1 -> gelu -> conv2 -> gelu -> maxpool2 -> conv3 -> gelu
//           -> mean over time  ==> feat[n_seq][256] (f32)
// One workgroup = 16 sequences; 8 waves of 32; dynamic LDS 111872 B.
// Padded LDS layouts (zero guard rows per sequence):
//   xs  [s][22]        f32   (interior l+1, pads at 0 and 21)
//   h1p [s][22][64]    f16
//   h2p [s][12][128]   f16   (interior lp+1, pads at 0 and 11)
// ---------------------------------------------------------------------------
__global__ __launch_bounds__(256)
void encoder_kernel(const float* __restrict__ x,
                    const float* __restrict__ w1, const float* __restrict__ b1,
                    const float* __restrict__ b2, const float* __restrict__ b3,
                    const h16* __restrict__ w2h, const h16* __restrict__ w3h,
                    float* __restrict__ feat_g) {
    extern __shared__ char smem[];
    float* feats = (float*)smem;               // 16*256 f32      = 16384 B
    float* xs    = (float*)(smem + 16384);     // 16*22  f32      =  1408 B
    h16*   h1p   = (h16*)(smem + 17792);       // 16*22*64  f16   = 45056 B
    h16*   h2p   = (h16*)(smem + 62848);       // 16*12*128 f16   = 49152 B

    const int tid  = threadIdx.x;
    const int wave = tid >> 5;
    const int lane = tid & 31;
    const int hlf16 = (lane >> 4) << 4;        // 16 * (lane/16), B-frag K offset
    const long long seq0 = (long long)blockIdx.x * 16;

    // Warm caches for the weight streams (global_prefetch_b8).
    __builtin_prefetch(w2h + tid * 96, 0, 0);
    __builtin_prefetch(w3h + tid * 384, 0, 0);

    // ---- phase 0: clear accumulators + pad rows, stage x -------------------
    for (int i = tid; i < 4096; i += 256) feats[i] = 0.0f;
    if (tid < 32)  xs[(tid >> 1) * 22 + (tid & 1) * 21] = 0.0f;          // x pads
    for (int i = tid; i < 2048; i += 256) {                              // h1p pads
        int s = i >> 7, r = (i >> 6) & 1, ci = i & 63;
        h1p[(s * 22 + r * 21) * 64 + ci] = (h16)0.0f;
    }
    for (int i = tid; i < 4096; i += 256) {                              // h2p pads
        int s = i >> 8, r = (i >> 7) & 1, ci = i & 127;
        h2p[(s * 12 + r * 11) * 128 + ci] = (h16)0.0f;
    }
    if (tid < 320) {
        int s = tid / 20, l = tid - s * 20;
        xs[s * 22 + l + 1] = x[seq0 * 20 + tid];
    }
    __syncthreads();

    // ---- phase 1: conv1 (Cin=1) + gelu -> h1p[s][l+1][ci] ------------------
    for (int i = tid; i < 20480; i += 256) {
        int ci  = i & 63;
        int pos = i >> 6;            // s*20 + l
        int s   = pos / 20, l = pos - s * 20;
        float acc = b1[ci];
#pragma unroll
        for (int t = 0; t < 3; ++t)
            acc += xs[s * 22 + l + t] * w1[ci * 3 + t];
        h1p[(s * 22 + l + 1) * 64 + ci] = (h16)gelu_f(acc);
    }
    __syncthreads();

    // ---- phase 2: conv2 via WMMA (M=co 128, N=pos 320, K=64 x 3 taps) ------
    // Each wave owns one co tile (cobase = wave*16); A-fragments hoisted.
    {
        const int cobase = wave << 4;
        v16h a2[6];
#pragma unroll
        for (int tap = 0; tap < 3; ++tap)
#pragma unroll
            for (int kc = 0; kc < 2; ++kc)
                a2[tap * 2 + kc] = frag_a(w2h + tap * 64, 192, cobase, kc << 5);

        for (int nt = 0; nt < 20; ++nt) {
            int nbase = nt << 4;
            int m = nbase + (lane & 15);
            int s = m / 20, l = m - s * 20;
            const h16* brow = h1p + (s * 22 + l) * 64 + hlf16;  // tap 0 row (l-1 padded)
            v8f acc = {};
#pragma unroll
            for (int tap = 0; tap < 3; ++tap)
#pragma unroll
                for (int kc = 0; kc < 2; ++kc)
                    acc = wmma_f16(a2[tap * 2 + kc],
                                   frag_ld(brow + tap * 64 + (kc << 5)), acc);
            // gelu + maxpool(2) -> h2p
            int hh = (lane >> 4) << 3;
#pragma unroll
            for (int r = 0; r < 8; ++r) {
                int co  = cobase + r + hh;
                float v = gelu_f(acc[r] + b2[co]);
                float w = __shfl_xor(v, 1, 32);         // partner column (l xor 1)
                float mx = fmaxf(v, w);
                if ((lane & 1) == 0)
                    h2p[(s * 12 + (l >> 1) + 1) * 128 + co] = (h16)mx;
            }
        }
    }
    __syncthreads();

    // ---- phase 3: conv3 via WMMA (M=co 256, N=pos2 160, K=128 x 3 taps) ----
    // Each wave owns two co tiles; 12 A-fragments hoisted per co tile.
    for (int cot = 0; cot < 2; ++cot) {
        const int cobase = (wave + (cot << 3)) << 4;
        v16h a3[12];
#pragma unroll
        for (int tap = 0; tap < 3; ++tap)
#pragma unroll
            for (int kc = 0; kc < 4; ++kc)
                a3[tap * 4 + kc] = frag_a(w3h + tap * 128, 384, cobase, kc << 5);

        for (int nt = 0; nt < 10; ++nt) {
            int nbase = nt << 4;
            int m = nbase + (lane & 15);
            int s = m / 10, lp = m - s * 10;
            const h16* brow = h2p + (s * 12 + lp) * 128 + hlf16; // tap 0 row
            v8f acc = {};
#pragma unroll
            for (int tap = 0; tap < 3; ++tap)
#pragma unroll
                for (int kc = 0; kc < 4; ++kc)
                    acc = wmma_f16(a3[tap * 4 + kc],
                                   frag_ld(brow + tap * 128 + (kc << 5)), acc);
            // gelu + mean(/10) accumulated into feats
            int hh = (lane >> 4) << 3;
#pragma unroll
            for (int r = 0; r < 8; ++r) {
                int co  = cobase + r + hh;
                float v = gelu_f(acc[r] + b3[co]);
                atomicAdd(&feats[s * 256 + co], v * 0.1f);
            }
        }
    }
    __syncthreads();

    // ---- phase 4: flush feat ----------------------------------------------
    for (int i = tid; i < 4096; i += 256)
        feat_g[seq0 * 256 + i] = feats[i];
}

// ---------------------------------------------------------------------------
// Kernel 2: per-(b,p): channel-mix GEMM + gelu + LayerNorm(C) + proj GEMM.
// One workgroup per (b,p) (496 total); dynamic LDS 217088 B.
// ---------------------------------------------------------------------------
__global__ __launch_bounds__(256)
void mix_proj_kernel(const float* __restrict__ feat_g,
                     const h16* __restrict__ mixh, const float* __restrict__ mix_b,
                     const float* __restrict__ ln_w, const float* __restrict__ ln_b,
                     const h16* __restrict__ projh, const float* __restrict__ proj_b,
                     float* __restrict__ out) {
    extern __shared__ char smem[];
    h16*   t_lds = (h16*)smem;                 // [d=256][c pad 160] f16 = 81920 B
    float* m_lds = (float*)(smem + 81920);     // [d=256][o pad 132] f32 = 135168 B

    const int tid  = threadIdx.x;
    const int wave = tid >> 5;
    const int lane = tid & 31;
    const int b = blockIdx.x / 62;
    const int p = blockIdx.x % 62;

    __builtin_prefetch(projh + tid * 256, 0, 0);

    // ---- stage t[d][c] = feat[b][c][p][d] (K padded to 160 with zeros) -----
    for (int c = 0; c < 160; ++c) {
        float v = 0.0f;
        if (c < 129)
            v = feat_g[(((long long)b * 129 + c) * 62 + p) * 256 + tid];
        t_lds[tid * 160 + c] = (h16)v;
    }
    __syncthreads();

    // ---- mix GEMM: m[d,o] = sum_c t[d,c] * mix_w[o,c] ----------------------
    // Each wave owns two d tiles; A-fragments (from LDS) hoisted per d tile.
    for (int dt = 0; dt < 2; ++dt) {
        const int dbase = (wave + (dt << 3)) << 4;
        v16h am[5];
#pragma unroll
        for (int kc = 0; kc < 5; ++kc)
            am[kc] = frag_a(t_lds, 160, dbase, kc << 5);

        for (int ot = 0; ot < 9; ++ot) {
            int obase = ot << 4;
            v8f acc = {};
#pragma unroll
            for (int kc = 0; kc < 5; ++kc)
                acc = wmma_f16(am[kc], frag_bt(mixh, 160, obase, kc << 5), acc);
            int o = obase + (lane & 15);
            float bias = (o < 129) ? mix_b[o] : 0.0f;
            int hh = (lane >> 4) << 3;
#pragma unroll
            for (int r = 0; r < 8; ++r) {
                int d = dbase + r + hh;
                float v = gelu_f(acc[r] + bias);
                if (o < 129) m_lds[d * 132 + o] = v;
            }
        }
    }
    __syncthreads();

    // ---- LayerNorm over o (C=129) per d; emit mixed[c][d] f16 (alias t_lds)-
    h16* mixed = t_lds;                    // [c pad 144][d=256] = 73728 B
    {
        int d = tid;
        float sum = 0.0f, ss = 0.0f;
        for (int o = 0; o < 129; ++o) {
            float v = m_lds[d * 132 + o];
            sum += v; ss += v * v;
        }
        float mu  = sum * (1.0f / 129.0f);
        float var = ss * (1.0f / 129.0f) - mu * mu;
        float rs  = rsqrtf(var + 1e-5f);
        for (int o = 0; o < 129; ++o) {
            float v = (m_lds[d * 132 + o] - mu) * rs * ln_w[o] + ln_b[o];
            mixed[o * 256 + d] = (h16)v;
        }
        for (int o = 129; o < 144; ++o)    // zero the padded rows
            mixed[o * 256 + d] = (h16)0.0f;
    }
    __syncthreads();

    // ---- proj GEMM: out[c,e] = sum_d mixed[c,d] * proj_w[e,d] + proj_b[e] --
    // Each wave owns two e tiles; B-fragments (proj_w, global) hoisted.
    for (int et = 0; et < 2; ++et) {
        const int ebase = (wave + (et << 3)) << 4;
        v16h bp[8];
#pragma unroll
        for (int kc = 0; kc < 8; ++kc)
            bp[kc] = frag_bt(projh, 256, ebase, kc << 5);

        int e  = ebase + (lane & 15);
        float pb = proj_b[e];

        for (int ct = 0; ct < 9; ++ct) {
            int cbase = ct << 4;
            v8f acc = {};
#pragma unroll
            for (int kc = 0; kc < 8; ++kc)
                acc = wmma_f16(frag_a(mixed, 256, cbase, kc << 5), bp[kc], acc);
            int hh = (lane >> 4) << 3;
#pragma unroll
            for (int r = 0; r < 8; ++r) {
                int c = cbase + r + hh;
                if (c < 129)
                    out[(((long long)b * 129 + c) * 62 + p) * 256 + e] = acc[r] + pb;
            }
        }
    }
}

// ---------------------------------------------------------------------------
// Host-side launch
// ---------------------------------------------------------------------------
extern "C" void kernel_launch(void* const* d_in, const int* in_sizes, int n_in,
                              void* d_out, int out_size, void* d_ws, size_t ws_size,
                              hipStream_t stream) {
    (void)in_sizes; (void)n_in; (void)out_size; (void)ws_size;

    const float* x      = (const float*)d_in[0];
    const float* w1     = (const float*)d_in[1];
    const float* b1     = (const float*)d_in[2];
    const float* w2     = (const float*)d_in[3];
    const float* b2     = (const float*)d_in[4];
    const float* w3     = (const float*)d_in[5];
    const float* b3     = (const float*)d_in[6];
    const float* mix_w  = (const float*)d_in[7];
    const float* mix_b  = (const float*)d_in[8];
    const float* ln_w   = (const float*)d_in[9];
    const float* ln_b   = (const float*)d_in[10];
    const float* proj_w = (const float*)d_in[11];
    const float* proj_b = (const float*)d_in[12];
    float* out = (float*)d_out;

    char* ws = (char*)d_ws;
    h16*   w2h    = (h16*)(ws + 0);          //  49152 B
    h16*   w3h    = (h16*)(ws + 49152);      // 196608 B
    h16*   mixh   = (h16*)(ws + 245760);     //  46080 B
    h16*   projh  = (h16*)(ws + 291840);     // 131072 B
    float* feat_g = (float*)(ws + 422912);   // 63984*256*4 = 65519616 B

    // 211456 conversion elements / 256
    prep_kernel<<<826, 256, 0, stream>>>(w2, w3, mix_w, proj_w,
                                         w2h, w3h, mixh, projh);

    // 63984 sequences / 16 per WG = 3999 workgroups; 111872 B dynamic LDS
    encoder_kernel<<<3999, 256, 111872, stream>>>(x, w1, b1, b2, b3,
                                                  w2h, w3h, feat_g);

    // B*P = 496 workgroups; 217088 B dynamic LDS
    mix_proj_kernel<<<496, 256, 217088, stream>>>(feat_g, mixh, mix_b,
                                                  ln_w, ln_b, projh, proj_b, out);
}
